// DifferentialAttention_2525440770143
// MI455X (gfx1250) — matrix operand order repaired and checked by reading, not compiled, not run on hardware
//
#include <hip/hip_runtime.h>
#include <hip/hip_bf16.h>
#include <stdint.h>

// ---------------------------------------------------------------------------
// Types for CDNA5 WMMA (wave32, 16x16x32 bf16)
// ---------------------------------------------------------------------------
typedef unsigned short ushort_t;
typedef __attribute__((ext_vector_type(16))) __bf16 v16bf;
typedef __attribute__((ext_vector_type(8)))  float  v8f;

union FragBF { uint4 u[2]; v16bf v; };

__device__ __forceinline__ ushort_t f2bf(float f) {
  uint32_t u = __float_as_uint(f);
  u += 0x7FFFu + ((u >> 16) & 1u);   // round-to-nearest-even
  return (ushort_t)(u >> 16);
}

// Problem constants
static constexpr int Bc  = 2;
static constexpr int Tc  = 2048;
static constexpr int Dc  = 1024;
static constexpr int Hc  = 16;
static constexpr int DHc = 64;
static constexpr int QKD = 2 * Hc * DHc;   // 2048

// ---------------------------------------------------------------------------
// 1) BitLinear weight quantization: W[f32, rows x in] -> bf16 (wq * scale)
// ---------------------------------------------------------------------------
__global__ void da_quant_kernel(const float* __restrict__ W,
                                ushort_t* __restrict__ Wq, int in_dim) {
  const int row = blockIdx.x;
  const float* wr = W + (size_t)row * in_dim;
  __shared__ float red[256];
  float s = 0.f;
  for (int c = threadIdx.x; c < in_dim; c += 256) s += fabsf(wr[c]);
  red[threadIdx.x] = s;
  __syncthreads();
  for (int off = 128; off > 0; off >>= 1) {
    if (threadIdx.x < off) red[threadIdx.x] += red[threadIdx.x + off];
    __syncthreads();
  }
  float scale = fmaxf(red[0] / (float)in_dim, 1e-5f);
  ushort_t* dst = Wq + (size_t)row * in_dim;
  for (int c = threadIdx.x; c < in_dim; c += 256) {
    float q = rintf(wr[c] / scale);
    q = fminf(fmaxf(q, -1.f), 1.f);
    dst[c] = f2bf(q * scale);
  }
}

// ---------------------------------------------------------------------------
// 2) x f32 -> bf16
// ---------------------------------------------------------------------------
__global__ void da_tobf16_kernel(const float* __restrict__ x,
                                 ushort_t* __restrict__ xb, int n) {
  int i = blockIdx.x * 256 + threadIdx.x;
  if (i < n) xb[i] = f2bf(x[i]);
}

// ---------------------------------------------------------------------------
// 3) lambda scalar
// ---------------------------------------------------------------------------
__global__ void da_lambda_kernel(const float* __restrict__ lq,
                                 const float* __restrict__ lk,
                                 float* __restrict__ lam, int n) {
  __shared__ float r1[256], r2[256];
  float a = 0.f, b = 0.f;
  for (int i = threadIdx.x; i < n; i += 256) { a += lq[i]; b += lk[i]; }
  r1[threadIdx.x] = a; r2[threadIdx.x] = b;
  __syncthreads();
  for (int off = 128; off > 0; off >>= 1) {
    if (threadIdx.x < off) { r1[threadIdx.x] += r1[threadIdx.x + off];
                             r2[threadIdx.x] += r2[threadIdx.x + off]; }
    __syncthreads();
  }
  if (threadIdx.x == 0) {
    float l = __expf((r1[0] - r2[0]) / (float)n);
    *lam = fminf(fmaxf(l, 0.1f), 2.0f);
  }
}

// ---------------------------------------------------------------------------
// 4) Generic bf16 WMMA GEMM: C[M,N] = X[M,K] * W[N,K]^T   (f32 out)
//    One wave per 32x64 C tile (two A fragments share every B fragment:
//    8 WMMAs per 12 b128 loads per k-step -> ~21 flop/byte of VMEM traffic).
//    A frag layout: lane row = lane&15; 16 contiguous halves at k0 + hi*16.
//    B frag layout: lane col = lane&15; 16 contiguous halves at k0 + hi*16.
// ---------------------------------------------------------------------------
__global__ void __launch_bounds__(32)
da_gemm_bf16_kernel(const ushort_t* __restrict__ X,
                    const ushort_t* __restrict__ W,
                    float* __restrict__ C, int M, int N, int K) {
  const int lane = threadIdx.x;
  const int r  = lane & 15;
  const int hi = lane >> 4;
  const int row0 = blockIdx.y * 32;
  const int col0 = blockIdx.x * 64;

  v8f accL0 = {}, accL1 = {}, accL2 = {}, accL3 = {};
  v8f accH0 = {}, accH1 = {}, accH2 = {}, accH3 = {};
  const ushort_t* xrowL = X + (size_t)(row0 + r) * K;
  const ushort_t* xrowH = X + (size_t)(row0 + 16 + r) * K;

  for (int k0 = 0; k0 < K; k0 += 32) {
    FragBF aL, aH;
    aL.u[0] = *(const uint4*)(xrowL + k0 + hi * 16);
    aL.u[1] = *(const uint4*)(xrowL + k0 + 8 + hi * 16);
    aH.u[0] = *(const uint4*)(xrowH + k0 + hi * 16);
    aH.u[1] = *(const uint4*)(xrowH + k0 + 8 + hi * 16);

    const ushort_t* w0 = W + (size_t)(col0 +  0 + r) * K + k0 + hi * 16;
    const ushort_t* w1 = W + (size_t)(col0 + 16 + r) * K + k0 + hi * 16;
    const ushort_t* w2 = W + (size_t)(col0 + 32 + r) * K + k0 + hi * 16;
    const ushort_t* w3 = W + (size_t)(col0 + 48 + r) * K + k0 + hi * 16;
    FragBF b0, b1, b2, b3;
    b0.u[0] = *(const uint4*)(w0); b0.u[1] = *(const uint4*)(w0 + 8);
    b1.u[0] = *(const uint4*)(w1); b1.u[1] = *(const uint4*)(w1 + 8);
    b2.u[0] = *(const uint4*)(w2); b2.u[1] = *(const uint4*)(w2 + 8);
    b3.u[0] = *(const uint4*)(w3); b3.u[1] = *(const uint4*)(w3 + 8);

    accL0 = __builtin_amdgcn_wmma_f32_16x16x32_bf16(false, aL.v, false, b0.v, (short)0, accL0, false, false);
    accH0 = __builtin_amdgcn_wmma_f32_16x16x32_bf16(false, aH.v, false, b0.v, (short)0, accH0, false, false);
    accL1 = __builtin_amdgcn_wmma_f32_16x16x32_bf16(false, aL.v, false, b1.v, (short)0, accL1, false, false);
    accH1 = __builtin_amdgcn_wmma_f32_16x16x32_bf16(false, aH.v, false, b1.v, (short)0, accH1, false, false);
    accL2 = __builtin_amdgcn_wmma_f32_16x16x32_bf16(false, aL.v, false, b2.v, (short)0, accL2, false, false);
    accH2 = __builtin_amdgcn_wmma_f32_16x16x32_bf16(false, aH.v, false, b2.v, (short)0, accH2, false, false);
    accL3 = __builtin_amdgcn_wmma_f32_16x16x32_bf16(false, aL.v, false, b3.v, (short)0, accL3, false, false);
    accH3 = __builtin_amdgcn_wmma_f32_16x16x32_bf16(false, aH.v, false, b3.v, (short)0, accH3, false, false);
  }

  // C/D layout: VGPR i -> row (i + 8*hi) within each 16-row half
  v8f* accsL[4] = { &accL0, &accL1, &accL2, &accL3 };
  v8f* accsH[4] = { &accH0, &accH1, &accH2, &accH3 };
#pragma unroll
  for (int n = 0; n < 4; ++n) {
#pragma unroll
    for (int i = 0; i < 8; ++i) {
      int growL = row0 + i + 8 * hi;
      int growH = row0 + 16 + i + 8 * hi;
      C[(size_t)growL * N + col0 + n * 16 + r] = (*accsL[n])[i];
      C[(size_t)growH * N + col0 + n * 16 + r] = (*accsH[n])[i];
    }
  }
}

// ---------------------------------------------------------------------------
// 5) LayerNorm over 2048-wide rows; output bf16 in [B, 2, H, T, DH]
// ---------------------------------------------------------------------------
__global__ void da_ln_kernel(const float* __restrict__ Cin,
                             const float* __restrict__ g,
                             const float* __restrict__ be,
                             ushort_t* __restrict__ dst) {
  const int rowi = blockIdx.x;          // [0, B*T)
  const int b = rowi / Tc, t = rowi % Tc;
  const float* x = Cin + (size_t)rowi * QKD;
  __shared__ float rs[256], rq[256];
  float s = 0.f, s2 = 0.f;
  for (int c = threadIdx.x; c < QKD; c += 256) { float v = x[c]; s += v; s2 += v * v; }
  rs[threadIdx.x] = s; rq[threadIdx.x] = s2;
  __syncthreads();
  for (int off = 128; off > 0; off >>= 1) {
    if (threadIdx.x < off) { rs[threadIdx.x] += rs[threadIdx.x + off];
                             rq[threadIdx.x] += rq[threadIdx.x + off]; }
    __syncthreads();
  }
  float mu  = rs[0] / (float)QKD;
  float var = rq[0] / (float)QKD - mu * mu;
  float inv = rsqrtf(var + 1e-5f);
  for (int c = threadIdx.x; c < QKD; c += 256) {
    float v = (x[c] - mu) * inv * g[c] + be[c];
    int br = c >> 10, rem = c & 1023, h = rem >> 6, dh = rem & 63;
    dst[((((size_t)b * 2 + br) * Hc + h) * Tc + t) * DHc + dh] = f2bf(v);
  }
}

// ---------------------------------------------------------------------------
// 6) V rearrange: Cv f32 [B*T, H*DH] -> bf16 vT [B, H, DH, T]
// ---------------------------------------------------------------------------
__global__ void da_vtr_kernel(const float* __restrict__ Cv,
                              ushort_t* __restrict__ vT) {
  int idx = blockIdx.x * 256 + threadIdx.x;   // over B*T*1024
  int col = idx & 1023, rowi = idx >> 10;
  int b = rowi / Tc, t = rowi % Tc;
  int h = col >> 6, dh = col & 63;
  vT[(((size_t)b * Hc + h) * DHc + dh) * Tc + t] = f2bf(Cv[idx]);
}

// ---------------------------------------------------------------------------
// 7) Flash attention (one branch per block). One wave handles 16 q rows,
//    streams 32-key tiles, WMMA for QK^T and P*V, online softmax.
//    q/k: [B,2,H,T,DH] bf16; vT: [B,H,DH,T] bf16; out: f32 [B,2,T,H*DH]
// ---------------------------------------------------------------------------
__global__ void __launch_bounds__(32)
da_attn_kernel(const ushort_t* __restrict__ qb,
               const ushort_t* __restrict__ kb,
               const ushort_t* __restrict__ vT,
               float* __restrict__ ob) {
  const int lane = threadIdx.x;
  const int r  = lane & 15;
  const int hi = lane >> 4;
  const int b  = blockIdx.z;
  const int br = blockIdx.y >> 4;     // / H (H==16)
  const int h  = blockIdx.y & 15;
  const int qt = blockIdx.x * 16;

  const size_t headoff = (((size_t)b * 2 + br) * Hc + h) * (size_t)Tc * DHc;
  const ushort_t* qp = qb + headoff + (size_t)qt * DHc;
  const ushort_t* kp = kb + headoff;
  const ushort_t* vp = vT + ((size_t)b * Hc + h) * (size_t)DHc * Tc;

  // Q A-fragments (dh 0..31 and 32..63)
  FragBF a0, a1;
  const ushort_t* qrow = qp + r * DHc;
  a0.u[0] = *(const uint4*)(qrow + hi * 16);
  a0.u[1] = *(const uint4*)(qrow + 8 + hi * 16);
  a1.u[0] = *(const uint4*)(qrow + 32 + hi * 16);
  a1.u[1] = *(const uint4*)(qrow + 40 + hi * 16);

  v8f o0 = {}, o1 = {}, o2 = {}, o3 = {};
  float m[8], l[8];
#pragma unroll
  for (int i = 0; i < 8; ++i) { m[i] = -1e30f; l[i] = 0.f; }

  __shared__ ushort_t pst[16][32];

  for (int kt = 0; kt < Tc; kt += 32) {
    v8f s0 = {}, s1 = {};
    {
      const ushort_t* kr = kp + (size_t)(kt + r) * DHc + hi * 16;
      FragBF kb0, kb1;
      kb0.u[0] = *(const uint4*)(kr);       kb0.u[1] = *(const uint4*)(kr + 8);
      kb1.u[0] = *(const uint4*)(kr + 32);  kb1.u[1] = *(const uint4*)(kr + 40);
      s0 = __builtin_amdgcn_wmma_f32_16x16x32_bf16(false, a0.v, false, kb0.v, (short)0, s0, false, false);
      s0 = __builtin_amdgcn_wmma_f32_16x16x32_bf16(false, a1.v, false, kb1.v, (short)0, s0, false, false);
    }
    {
      const ushort_t* kr = kp + (size_t)(kt + 16 + r) * DHc + hi * 16;
      FragBF kb0, kb1;
      kb0.u[0] = *(const uint4*)(kr);       kb0.u[1] = *(const uint4*)(kr + 8);
      kb1.u[0] = *(const uint4*)(kr + 32);  kb1.u[1] = *(const uint4*)(kr + 40);
      s1 = __builtin_amdgcn_wmma_f32_16x16x32_bf16(false, a0.v, false, kb0.v, (short)0, s1, false, false);
      s1 = __builtin_amdgcn_wmma_f32_16x16x32_bf16(false, a1.v, false, kb1.v, (short)0, s1, false, false);
    }

    // online softmax: VGPR i <-> row (i + 8*hi); 16 cols spread over a lane-half
    float alpha[8];
#pragma unroll
    for (int i = 0; i < 8; ++i) {
      float v0 = s0[i] * 0.125f, v1 = s1[i] * 0.125f;   // DH^-0.5
      s0[i] = v0; s1[i] = v1;
      float mx = fmaxf(v0, v1);
      mx = fmaxf(mx, __shfl_xor(mx, 1, 32));
      mx = fmaxf(mx, __shfl_xor(mx, 2, 32));
      mx = fmaxf(mx, __shfl_xor(mx, 4, 32));
      mx = fmaxf(mx, __shfl_xor(mx, 8, 32));
      float mn = fmaxf(m[i], mx);
      alpha[i] = __expf(m[i] - mn);
      m[i] = mn;
    }
#pragma unroll
    for (int i = 0; i < 8; ++i) {
      float p0 = __expf(s0[i] - m[i]);
      float p1 = __expf(s1[i] - m[i]);
      float sum = p0 + p1;
      sum += __shfl_xor(sum, 1, 32);
      sum += __shfl_xor(sum, 2, 32);
      sum += __shfl_xor(sum, 4, 32);
      sum += __shfl_xor(sum, 8, 32);
      l[i] = l[i] * alpha[i] + sum;
      int row = i + 8 * hi;
      pst[row][r]      = f2bf(p0);
      pst[row][16 + r] = f2bf(p1);
      o0[i] *= alpha[i]; o1[i] *= alpha[i]; o2[i] *= alpha[i]; o3[i] *= alpha[i];
    }

    // P from LDS in A-layout (DS ops are in-order within a wave)
    FragBF pa;
    pa.u[0] = *(const uint4*)(&pst[r][hi * 16]);
    pa.u[1] = *(const uint4*)(&pst[r][8 + hi * 16]);

    // V B-fragments: dh = n*16 + r, 16 contiguous keys at kt + hi*16
    const ushort_t* v0p = vp + (size_t)( 0 + r) * Tc + kt + hi * 16;
    const ushort_t* v1p = vp + (size_t)(16 + r) * Tc + kt + hi * 16;
    const ushort_t* v2p = vp + (size_t)(32 + r) * Tc + kt + hi * 16;
    const ushort_t* v3p = vp + (size_t)(48 + r) * Tc + kt + hi * 16;
    FragBF vb0, vb1, vb2, vb3;
    vb0.u[0] = *(const uint4*)(v0p); vb0.u[1] = *(const uint4*)(v0p + 8);
    vb1.u[0] = *(const uint4*)(v1p); vb1.u[1] = *(const uint4*)(v1p + 8);
    vb2.u[0] = *(const uint4*)(v2p); vb2.u[1] = *(const uint4*)(v2p + 8);
    vb3.u[0] = *(const uint4*)(v3p); vb3.u[1] = *(const uint4*)(v3p + 8);

    o0 = __builtin_amdgcn_wmma_f32_16x16x32_bf16(false, pa.v, false, vb0.v, (short)0, o0, false, false);
    o1 = __builtin_amdgcn_wmma_f32_16x16x32_bf16(false, pa.v, false, vb1.v, (short)0, o1, false, false);
    o2 = __builtin_amdgcn_wmma_f32_16x16x32_bf16(false, pa.v, false, vb2.v, (short)0, o2, false, false);
    o3 = __builtin_amdgcn_wmma_f32_16x16x32_bf16(false, pa.v, false, vb3.v, (short)0, o3, false, false);
  }

  const size_t obase = (((size_t)b * 2 + br) * Tc + qt) * (size_t)(Hc * DHc);
  v8f* os[4] = { &o0, &o1, &o2, &o3 };
#pragma unroll
  for (int n = 0; n < 4; ++n) {
#pragma unroll
    for (int i = 0; i < 8; ++i) {
      int grow = i + 8 * hi;
      ob[obase + (size_t)grow * (Hc * DHc) + h * DHc + n * 16 + r] = (*os[n])[i] / l[i];
    }
  }
}

// ---------------------------------------------------------------------------
// 8) combine: z = bf16(out1 - lam*out2), [B*T, 1024]
// ---------------------------------------------------------------------------
__global__ void da_combine_kernel(const float* __restrict__ ob,
                                  const float* __restrict__ lam,
                                  ushort_t* __restrict__ z) {
  int idx = blockIdx.x * 256 + threadIdx.x;   // B*T*1024
  int col = idx & 1023, rowi = idx >> 10;
  int b = rowi / Tc, t = rowi % Tc;
  float lv = *lam;
  size_t base = (size_t)b * 2 * Tc * 1024;
  float v1 = ob[base + (size_t)t * 1024 + col];
  float v2 = ob[base + (size_t)Tc * 1024 + (size_t)t * 1024 + col];
  z[idx] = f2bf(v1 - lv * v2);
}

// ---------------------------------------------------------------------------
// Launch
// ---------------------------------------------------------------------------
extern "C" void kernel_launch(void* const* d_in, const int* in_sizes, int n_in,
                              void* d_out, int out_size, void* d_ws, size_t ws_size,
                              hipStream_t stream) {
  const float* x   = (const float*)d_in[0];
  const float* Wq  = (const float*)d_in[1];
  const float* Wk  = (const float*)d_in[2];
  const float* Wv  = (const float*)d_in[3];
  const float* Wo  = (const float*)d_in[4];
  const float* lq  = (const float*)d_in[5];
  const float* lk  = (const float*)d_in[6];
  const float* qg  = (const float*)d_in[7];
  const float* qbt = (const float*)d_in[8];
  const float* kg  = (const float*)d_in[9];
  const float* kbt = (const float*)d_in[10];

  char* ws = (char*)d_ws;
  size_t off = 0;
  auto alloc = [&](size_t bytes) { char* p = ws + off; off += (bytes + 255) & ~(size_t)255; return p; };

  const size_t MT = (size_t)Bc * Tc;                 // 4096 rows
  ushort_t* wqq = (ushort_t*)alloc((size_t)QKD * Dc * 2);
  ushort_t* wkq = (ushort_t*)alloc((size_t)QKD * Dc * 2);
  ushort_t* wvq = (ushort_t*)alloc((size_t)Dc * Dc * 2);
  ushort_t* woq = (ushort_t*)alloc((size_t)Dc * Dc * 2);
  ushort_t* xbf = (ushort_t*)alloc(MT * Dc * 2);
  float*    Cq  = (float*)   alloc(MT * QKD * 4);    // reused as attn-out (both branches)
  float*    Ck  = (float*)   alloc(MT * QKD * 4);    // (contiguous with Cq)
  float*    Cv  = (float*)   alloc(MT * Dc * 4);     // reused as z
  ushort_t* qbf = (ushort_t*)alloc((size_t)Bc * 2 * Hc * Tc * DHc * 2);
  ushort_t* kbf = (ushort_t*)alloc((size_t)Bc * 2 * Hc * Tc * DHc * 2);
  ushort_t* vT  = (ushort_t*)alloc((size_t)Bc * Hc * DHc * Tc * 2);
  float*    lam = (float*)   alloc(256);
  (void)Ck; (void)ws_size; (void)n_in; (void)in_sizes; (void)out_size;

  float*    obuf = Cq;            // [B,2,T,1024] f32 = 32MB, fits in Cq
  ushort_t* z    = (ushort_t*)Cv; // [B*T,1024] bf16 over Cv

  // 1) quantize weights
  da_quant_kernel<<<QKD, 256, 0, stream>>>(Wq, wqq, Dc);
  da_quant_kernel<<<QKD, 256, 0, stream>>>(Wk, wkq, Dc);
  da_quant_kernel<<<Dc, 256, 0, stream>>>(Wv, wvq, Dc);
  da_quant_kernel<<<Dc, 256, 0, stream>>>(Wo, woq, Dc);

  // 2) x -> bf16
  {
    int n = (int)(MT * Dc);
    da_tobf16_kernel<<<(n + 255) / 256, 256, 0, stream>>>(x, xbf, n);
  }

  // 3) lambda
  da_lambda_kernel<<<1, 256, 0, stream>>>(lq, lk, lam, Hc * DHc);

  // 4) projections (WMMA GEMM, 32x64 tiles per wave)
  da_gemm_bf16_kernel<<<dim3(QKD / 64, (int)MT / 32), 32, 0, stream>>>(xbf, wqq, Cq, (int)MT, QKD, Dc);
  da_gemm_bf16_kernel<<<dim3(QKD / 64, (int)MT / 32), 32, 0, stream>>>(xbf, wkq, Ck, (int)MT, QKD, Dc);
  da_gemm_bf16_kernel<<<dim3(Dc  / 64, (int)MT / 32), 32, 0, stream>>>(xbf, wvq, Cv, (int)MT, Dc, Dc);

  // 5) layernorm -> [B,2,H,T,DH] bf16
  da_ln_kernel<<<(int)MT, 256, 0, stream>>>(Cq, qg, qbt, qbf);
  da_ln_kernel<<<(int)MT, 256, 0, stream>>>(Ck, kg, kbt, kbf);

  // 6) V -> [B,H,DH,T] bf16
  da_vtr_kernel<<<(int)(MT * Dc) / 256, 256, 0, stream>>>(Cv, vT);

  // 7) attention (Cq/Ck now dead -> obuf aliases Cq)
  da_attn_kernel<<<dim3(Tc / 16, 2 * Hc, Bc), 32, 0, stream>>>(qbf, kbf, vT, obuf);

  // 8) combine (Cv now dead -> z aliases it)
  da_combine_kernel<<<(int)(MT * Dc) / 256, 256, 0, stream>>>(obuf, lam, z);

  // 9) output projection -> d_out (f32)
  da_gemm_bf16_kernel<<<dim3(Dc / 64, (int)MT / 32), 32, 0, stream>>>(z, woq, (float*)d_out, (int)MT, Dc, Dc);
}
